// LSTMCell_19353122636363
// MI455X (gfx1250) — matrix operand order repaired and checked by reading, not compiled, not run on hardware
//
#include <hip/hip_runtime.h>
#include <math.h>

// ---------------------------------------------------------------------------
// LayerNorm-LSTM cell for MI455X (gfx1250, wave32, WMMA).
// Phase 1: v = x@W_ih^T + h@W_hh^T  via bf16x3 split-precision WMMA (fp32-class
//          accuracy), v (4096x4096 f32, 64MB) staged in d_ws.
// Phase 2: per-row layernorm of 4 gates + LSTM update + layernorm(new_c).
//
// Round-2 change: all WMMA fragments for a K-chunk are preloaded from LDS
// before the WMMA chain, so ds_load -> consumer distance is maximized and
// s_wait_dscnt can retire at nonzero counts instead of serializing per tile.
// ---------------------------------------------------------------------------

typedef __attribute__((ext_vector_type(16))) __bf16        v16bf;
typedef __attribute__((ext_vector_type(8)))  float         v8f;
typedef __attribute__((ext_vector_type(4)))  float         f32x4;
typedef __attribute__((ext_vector_type(4)))  unsigned int  u32x4;

#define TM   128   // block tile M (rows of x/h)
#define TN   128   // block tile N (rows of W = output cols)
#define KC   32    // K chunk per iteration (one bf16 WMMA K)
#define LSTR 40    // LDS row stride in bf16 elems (80B: 16B-aligned, padded)

#define BDIM 4096  // batch
#define NDIM 4096  // 4*H
#define KDIM 1024  // D == H

__global__ __launch_bounds__(256)
void gemm_gates_wmma(const float* __restrict__ x, const float* __restrict__ h,
                     const float* __restrict__ Wih, const float* __restrict__ Whh,
                     float* __restrict__ vout)
{
    __shared__ alignas(16) __bf16 Ahi[TM][LSTR];
    __shared__ alignas(16) __bf16 Alo[TM][LSTR];
    __shared__ alignas(16) __bf16 Bhi[TN][LSTR];
    __shared__ alignas(16) __bf16 Blo[TN][LSTR];

    const int tid  = threadIdx.x;
    const int lane = tid & 31;
    const int wv   = tid >> 5;          // wave 0..7
    const int bM   = blockIdx.y * TM;
    const int bN   = blockIdx.x * TN;
    const int waveM = (wv & 3) * 32;    // 4 waves along M
    const int waveN = (wv >> 2) * 64;   // 2 waves along N

    // global tile loaders: 256 threads * 4 passes cover 128 rows x 32 cols f32
    const int lrow = tid >> 3;          // 0..31
    const int lcol = (tid & 7) * 4;     // 0..28 step 4

    v8f acc[2][4];
    const v8f vzero = {0.f,0.f,0.f,0.f,0.f,0.f,0.f,0.f};
#pragma unroll
    for (int i = 0; i < 2; ++i)
#pragma unroll
        for (int j = 0; j < 4; ++j) acc[i][j] = vzero;

    f32x4 areg[4], breg[4];

    auto gload = [&](int it) {
        const float* As = (it < 32) ? x   : h;    // A source (rows = batch)
        const float* Ws = (it < 32) ? Wih : Whh;  // B source (rows = out cols)
        const int kc = (it & 31) * KC;
#pragma unroll
        for (int p = 0; p < 4; ++p) {
            areg[p] = *(const f32x4*)&As[(size_t)(bM + lrow + p*32) * KDIM + kc + lcol];
            breg[p] = *(const f32x4*)&Ws[(size_t)(bN + lrow + p*32) * KDIM + kc + lcol];
        }
    };

    auto lstore = [&]() {
#pragma unroll
        for (int p = 0; p < 4; ++p) {
            const float av[4] = {areg[p].x, areg[p].y, areg[p].z, areg[p].w};
            const float bv[4] = {breg[p].x, breg[p].y, breg[p].z, breg[p].w};
            const int r = lrow + p * 32;
#pragma unroll
            for (int j = 0; j < 4; ++j) {
                __bf16 ah = (__bf16)av[j];
                Ahi[r][lcol + j] = ah;
                Alo[r][lcol + j] = (__bf16)(av[j] - (float)ah);
                __bf16 bh = (__bf16)bv[j];
                Bhi[r][lcol + j] = bh;
                Blo[r][lcol + j] = (__bf16)(bv[j] - (float)bh);
            }
        }
    };

    union Frag { v16bf v; u32x4 q[2]; };

    // A fragment (16x32 bf16): lane<16 -> M=lane, K=0..7 & 16..23;
    //                          lane>=16 -> M=lane-16, K=8..15 & 24..31.
    auto loadA = [&](__bf16 (*S)[LSTR], int mt) -> v16bf {
        const int m  = waveM + mt * 16 + (lane & 15);
        const int k0 = (lane >> 4) * 8;
        Frag f;
        f.q[0] = *(const u32x4*)&S[m][k0];
        f.q[1] = *(const u32x4*)&S[m][k0 + 16];
        return f.v;
    };
    // B fragment (32x16 bf16): lane holds column N=lane%16,
    // K = (lane/16)*8 .. +7 and 16+(lane/16)*8 .. +7 (same order as A).
    auto loadB = [&](__bf16 (*S)[LSTR], int nt) -> v16bf {
        const int n  = waveN + nt * 16 + (lane & 15);
        const int k0 = (lane >> 4) * 8;
        Frag f;
        f.q[0] = *(const u32x4*)&S[n][k0];
        f.q[1] = *(const u32x4*)&S[n][k0 + 16];
        return f.v;
    };

    gload(0);
    for (int it = 0; it < 64; ++it) {
        __syncthreads();              // previous chunk's LDS reads done
        lstore();                     // convert fp32 -> bf16 hi/lo into LDS
        __syncthreads();

        // Preload ALL fragments for this chunk first (24 ds_load_b128),
        // so the WMMA chain below never waits on a just-issued LDS load.
        v16bf afh[2], afl[2], bfh[4], bfl[4];
#pragma unroll
        for (int mt = 0; mt < 2; ++mt) {
            afh[mt] = loadA(Ahi, mt);
            afl[mt] = loadA(Alo, mt);
        }
#pragma unroll
        for (int nt = 0; nt < 4; ++nt) {
            bfh[nt] = loadB(Bhi, nt);
            bfl[nt] = loadB(Blo, nt);
        }

        if (it + 1 < 64) gload(it + 1);   // global prefetch under WMMA chain

#pragma unroll
        for (int nt = 0; nt < 4; ++nt)
#pragma unroll
            for (int mt = 0; mt < 2; ++mt) {
                // bf16x3: hi*hi + lo*hi + hi*lo  (drop lo*lo ~2^-16 rel)
                acc[mt][nt] = __builtin_amdgcn_wmma_f32_16x16x32_bf16(
                    false, afh[mt], false, bfh[nt], (short)0, acc[mt][nt], false, false);
                acc[mt][nt] = __builtin_amdgcn_wmma_f32_16x16x32_bf16(
                    false, afl[mt], false, bfh[nt], (short)0, acc[mt][nt], false, false);
                acc[mt][nt] = __builtin_amdgcn_wmma_f32_16x16x32_bf16(
                    false, afh[mt], false, bfl[nt], (short)0, acc[mt][nt], false, false);
            }
    }

    // C/D layout: VGPR r -> lanes 0-15: (M=r, N=lane); lanes 16-31: (M=r+8).
#pragma unroll
    for (int mt = 0; mt < 2; ++mt)
#pragma unroll
        for (int nt = 0; nt < 4; ++nt) {
            const int m0 = bM + waveM + mt * 16 + (lane >> 4) * 8;
            const int n0 = bN + waveN + nt * 16 + (lane & 15);
#pragma unroll
            for (int r = 0; r < 8; ++r)
                vout[(size_t)(m0 + r) * NDIM + n0] = acc[mt][nt][r];
        }
}

// ---------------------------------------------------------------------------
// Epilogue: per-row gate layernorm + LSTM update + cell layernorm.
// One 256-thread block per batch row; 4 H-positions per thread.
// ---------------------------------------------------------------------------

__device__ inline float block_sum(float val, float* red)
{
    const int t = threadIdx.x;
    red[t] = val;
    __syncthreads();
    for (int st = 128; st > 0; st >>= 1) {
        if (t < st) red[t] += red[t + st];
        __syncthreads();
    }
    float r = red[0];
    __syncthreads();
    return r;
}

__device__ inline float sigmoidf(float v) { return 1.0f / (1.0f + expf(-v)); }

__global__ __launch_bounds__(256)
void lstm_epilogue(const float* __restrict__ v, const float* __restrict__ c,
                   const float* __restrict__ b_ih,
                   const float* __restrict__ gamma_ifgo, const float* __restrict__ beta_ifgo,
                   const float* __restrict__ gamma_c, const float* __restrict__ beta_c,
                   float* __restrict__ out_h, float* __restrict__ out_c)
{
    __shared__ float red[256];
    const int b  = blockIdx.x;
    const int t  = threadIdx.x;
    const int h0 = t * 4;

    float gv[4][4];
#pragma unroll
    for (int g = 0; g < 4; ++g) {
        f32x4 vv = *(const f32x4*)&v[(size_t)b * NDIM + g * 1024 + h0];
        float e[4] = {vv.x, vv.y, vv.z, vv.w};
#pragma unroll
        for (int j = 0; j < 4; ++j) e[j] += b_ih[g * 1024 + h0 + j];

        float mean = block_sum(e[0] + e[1] + e[2] + e[3], red) * (1.0f / 1024.0f);
        float q = 0.0f;
#pragma unroll
        for (int j = 0; j < 4; ++j) { float d = e[j] - mean; q += d * d; }
        float var = block_sum(q, red) * (1.0f / 1023.0f);   // unbiased
        float inv = 1.0f / (sqrtf(var) + 1e-6f);
#pragma unroll
        for (int j = 0; j < 4; ++j)
            gv[g][j] = gamma_ifgo[g * 1024 + h0 + j] * (e[j] - mean) * inv
                     + beta_ifgo[g * 1024 + h0 + j];
    }

    float nc[4];
#pragma unroll
    for (int j = 0; j < 4; ++j) {
        float ig = sigmoidf(gv[0][j]);
        float fg = sigmoidf(gv[1][j] + 1.0f);
        float gg = tanhf(gv[2][j]);
        nc[j] = c[(size_t)b * 1024 + h0 + j] * fg + ig * gg;
    }

    float mean = block_sum(nc[0] + nc[1] + nc[2] + nc[3], red) * (1.0f / 1024.0f);
    float q = 0.0f;
#pragma unroll
    for (int j = 0; j < 4; ++j) { float d = nc[j] - mean; q += d * d; }
    float var = block_sum(q, red) * (1.0f / 1023.0f);
    float inv = 1.0f / (sqrtf(var) + 1e-6f);

#pragma unroll
    for (int j = 0; j < 4; ++j) {
        float cl = gamma_c[h0 + j] * (nc[j] - mean) * inv + beta_c[h0 + j];
        out_c[(size_t)b * 1024 + h0 + j] = cl;
        out_h[(size_t)b * 1024 + h0 + j] = tanhf(cl) * sigmoidf(gv[3][j]);
    }
}

// ---------------------------------------------------------------------------

extern "C" void kernel_launch(void* const* d_in, const int* in_sizes, int n_in,
                              void* d_out, int out_size, void* d_ws, size_t ws_size,
                              hipStream_t stream)
{
    const float* x     = (const float*)d_in[0];
    const float* h     = (const float*)d_in[1];
    const float* c     = (const float*)d_in[2];
    const float* Wih   = (const float*)d_in[3];
    const float* bih   = (const float*)d_in[4];
    const float* Whh   = (const float*)d_in[5];
    const float* gifgo = (const float*)d_in[6];
    const float* bifgo = (const float*)d_in[7];
    const float* gc    = (const float*)d_in[8];
    const float* bc    = (const float*)d_in[9];

    float* out  = (float*)d_out;
    float* vtmp = (float*)d_ws;   // needs 4096*4096*4 = 64 MB scratch

    dim3 g1(NDIM / TN, BDIM / TM);   // 32 x 32 blocks
    gemm_gates_wmma<<<g1, 256, 0, stream>>>(x, h, Wih, Whh, vtmp);

    lstm_epilogue<<<BDIM, 256, 0, stream>>>(vtmp, c, bih, gifgo, bifgo, gc, bc,
                                            out, out + (size_t)BDIM * 1024);
}
